// Injector_53721450938958
// MI455X (gfx1250) — matrix operand order repaired and checked by reading, not compiled, not run on hardware
//
#include <hip/hip_runtime.h>
#include <hip/hip_bf16.h>

// ---------------- problem constants ----------------
#define BB   2
#define NQ   16384
#define CC   768
#define NHH  6
#define NPP  4
#define DD   128
#define HWW  128
#define ROWS (BB * NQ)          // 32768 query rows
#define EPSF 1e-6f

// GEMM tiling
#define BM   128                // block M tile
#define BN   128                // block N tile
#define BK   32                 // k step (one WMMA K)
#define LDSS 40                 // LDS row stride in halves (32 data + 8 pad, 80B: 16B aligned)

typedef __attribute__((ext_vector_type(16))) _Float16 v16h;
typedef __attribute__((ext_vector_type(8)))  _Float16 v8h;
typedef __attribute__((ext_vector_type(4)))  _Float16 v4h;
typedef __attribute__((ext_vector_type(8)))  float    v8f;

// ---------------- WMMA fragment loaders (CDNA5 16-bit layouts) ----------------
// A (16x32 f16, row-major MxK source):
//   lanes 0-15 : row M=lane,    VGPR0-3 = K 0..7,  VGPR4-7 = K 16..23
//   lanes 16-31: row M=lane-16, VGPR0-3 = K 8..15, VGPR4-7 = K 24..31
__device__ __forceinline__ v16h load_a_frag(const _Float16* __restrict__ A, int ldK,
                                            int mrow, int k0, int lane) {
  const int r  = mrow + (lane & 15);
  const int kb = k0 + ((lane >> 4) << 3);
  const _Float16* p = A + (size_t)r * ldK + kb;
  v8h lo = *(const v8h*)(p);        // K = kb .. kb+7
  v8h hi = *(const v8h*)(p + 16);   // K = kb+16 .. kb+23
  return __builtin_shufflevector(lo, hi, 0,1,2,3,4,5,6,7,8,9,10,11,12,13,14,15);
}

// B (32x16 f16, KxN): lanes 0-15 hold K=0..7,16..23 of col N=lane, lanes 16-31 hold K=8..15,24..31.
// LDS tile is stored column-major-in-N / K-contiguous, so each lane reads two contiguous 16B runs (K kb..kb+7, kb+16..kb+23).
__device__ __forceinline__ v16h lds_b_frag(const _Float16* lbuf, int nloc, int lane) {
  const int n  = nloc + (lane & 15);
  const int kb = (lane >> 4) << 3;
  const _Float16* p = lbuf + n * LDSS + kb;
  v8h lo = *(const v8h*)(p);
  v8h hi = *(const v8h*)(p + 16);
  return __builtin_shufflevector(lo, hi, 0,1,2,3,4,5,6,7,8,9,10,11,12,13,14,15);
}

// ---------------- async global->LDS tile copy (ASYNCcnt path) ----------------
// Copies a BN x BK f16 tile of Bt (N-major, ldK=K) into lbuf with row stride LDSS.
// 512 x 16B chunks; 256 threads x 2 chunks. Each lane issues its own 16B async copy.
__device__ __forceinline__ void async_tile_copy(const _Float16* __restrict__ Bt, int ldK,
                                                int ncol0, int k0, _Float16* lbuf) {
  const int t = threadIdx.x;
#pragma unroll
  for (int i = 0; i < 2; ++i) {
    const int e    = t + i * 256;          // 0..511
    const int n    = e >> 2;               // tile row (N)
    const int koff = (e & 3) << 3;         // 0,8,16,24 halves
    const _Float16* gp = Bt + (size_t)(ncol0 + n) * ldK + k0 + koff;
    const unsigned laddr = (unsigned)(uintptr_t)(lbuf + n * LDSS + koff);
    asm volatile("global_load_async_to_lds_b128 %0, %1, off"
                 :: "v"(laddr), "v"(gp) : "memory");
  }
}
__device__ __forceinline__ void wait_async() {
  asm volatile("s_wait_asynccnt 0x0" ::: "memory");
}

#define WMMA_F16(a, b, c) \
  __builtin_amdgcn_wmma_f32_16x16x32_f16(false, (a), false, (b), (short)0, (c), false, false)

// ---------------- kernel 1: weight transpose + f32->f16 convert ----------------
__global__ void __launch_bounds__(256)
wt_convert_kernel(const float* __restrict__ W, _Float16* __restrict__ Wt, int K, int N) {
  int idx = blockIdx.x * 256 + threadIdx.x;
  if (idx >= K * N) return;
  int k = idx / N, n = idx - k * N;
  Wt[(size_t)n * K + k] = (_Float16)W[(size_t)k * N + n];
}

// ---------------- kernel 2: layernorm (f32 in -> f16 out), one row per block ----------------
__global__ void __launch_bounds__(256)
layernorm_kernel(const float* __restrict__ x, const float* __restrict__ g,
                 const float* __restrict__ b, _Float16* __restrict__ y) {
  const int row = blockIdx.x;
  const float* xr = x + (size_t)row * CC;
  __shared__ float ssum[256], ssq[256];
  float s = 0.f, s2 = 0.f;
  for (int i = threadIdx.x; i < CC; i += 256) { float v = xr[i]; s += v; s2 += v * v; }
  ssum[threadIdx.x] = s; ssq[threadIdx.x] = s2;
  __syncthreads();
  for (int off = 128; off > 0; off >>= 1) {
    if (threadIdx.x < off) {
      ssum[threadIdx.x] += ssum[threadIdx.x + off];
      ssq[threadIdx.x]  += ssq[threadIdx.x + off];
    }
    __syncthreads();
  }
  const float mean = ssum[0] * (1.f / CC);
  const float var  = ssq[0] * (1.f / CC) - mean * mean;
  const float rs   = rsqrtf(var + EPSF);
  _Float16* yr = y + (size_t)row * CC;
  for (int i = threadIdx.x; i < CC; i += 256)
    yr[i] = (_Float16)((xr[i] - mean) * rs * g[i] + b[i]);
}

// ---------------- kernel 3: WMMA GEMM (value projection), f16 out + bias ----------------
// 256 thr = 8 waves as 4(M) x 2(N); wave tile 32x64; block tile 128x128.
// B staged in LDS via async copies, double buffered; A streamed from global.
__global__ void __launch_bounds__(256)
gemm_bias_f16_kernel(const _Float16* __restrict__ A, const _Float16* __restrict__ Bt,
                     const float* __restrict__ bias, _Float16* __restrict__ Y,
                     int K, int N) {
  __shared__ _Float16 lbs[2][BN * LDSS];
  const int lane  = threadIdx.x & 31;
  const int wave  = threadIdx.x >> 5;
  const int wm    = wave >> 1, wn = wave & 1;
  const int mrow  = blockIdx.x * BM + wm * 32;
  const int ncol0 = blockIdx.y * BN;
  const int nloc  = wn * 64;

  v8f acc[2][4] = {};
  async_tile_copy(Bt, K, ncol0, 0, lbs[0]);
  const int KT = K / BK;
  for (int t = 0; t < KT; ++t) {
    wait_async();
    __syncthreads();                       // whole tile t visible to all waves
    if (t + 1 < KT) async_tile_copy(Bt, K, ncol0, (t + 1) * BK, lbs[(t + 1) & 1]);
    const _Float16* lb = lbs[t & 1];
    const int k0 = t * BK;
    v16h a0 = load_a_frag(A, K, mrow,      k0, lane);
    v16h a1 = load_a_frag(A, K, mrow + 16, k0, lane);
#pragma unroll
    for (int j = 0; j < 4; ++j) {
      v16h b = lds_b_frag(lb, nloc + j * 16, lane);
      acc[0][j] = WMMA_F16(a0, b, acc[0][j]);
      acc[1][j] = WMMA_F16(a1, b, acc[1][j]);
    }
  }
  // C/D layout: VGPR r -> row (r + 8*(lane>=16)), col = lane&15
  const int c = lane & 15;
#pragma unroll
  for (int i = 0; i < 2; ++i) {
    const int rbase = mrow + i * 16 + ((lane >> 4) << 3);
#pragma unroll
    for (int j = 0; j < 4; ++j) {
      const int col = ncol0 + nloc + j * 16 + c;
      const float bs = bias[col];
#pragma unroll
      for (int r = 0; r < 8; ++r)
        Y[(size_t)(rbase + r) * N + col] = (_Float16)(acc[i][j][r] + bs);
    }
  }
}

// ---------------- kernel 4: offsets + attention logits + softmax + sample coords ----------------
__global__ void __launch_bounds__(256)
offset_attn_kernel(const _Float16* __restrict__ qln, const float* __restrict__ rp,
                   const float* __restrict__ Woff, const float* __restrict__ boff,
                   const float* __restrict__ Wattn, const float* __restrict__ battn,
                   float2* __restrict__ sloc, float* __restrict__ awout) {
  const int idx = blockIdx.x * 256 + threadIdx.x;        // (b*NQ+q)*NH + h
  if (idx >= ROWS * NHH) return;
  const int h  = idx % NHH;
  const int bq = idx / NHH;
  const _Float16* qp = qln + (size_t)bq * CC;
  float ao[8] = {0,0,0,0,0,0,0,0};
  float aa[4] = {0,0,0,0};
  for (int k = 0; k < CC; ++k) {
    const float ql = (float)qp[k];
    const float* wo = Woff  + (size_t)k * (NHH * NPP * 2) + h * (NPP * 2);
    const float* wa = Wattn + (size_t)k * (NHH * NPP)     + h * NPP;
#pragma unroll
    for (int j = 0; j < 8; ++j) ao[j] += ql * wo[j];
#pragma unroll
    for (int j = 0; j < 4; ++j) aa[j] += ql * wa[j];
  }
  float lg[4], mx = -3.4e38f;
#pragma unroll
  for (int j = 0; j < 4; ++j) { lg[j] = aa[j] + battn[h * NPP + j]; mx = fmaxf(mx, lg[j]); }
  float den = 0.f;
#pragma unroll
  for (int j = 0; j < 4; ++j) { lg[j] = __expf(lg[j] - mx); den += lg[j]; }
  const float inv = 1.f / den;
  const float rx = rp[(size_t)bq * 2 + 0];
  const float ry = rp[(size_t)bq * 2 + 1];
#pragma unroll
  for (int p = 0; p < 4; ++p) {
    const float ox = ao[2 * p + 0] + boff[h * (NPP * 2) + 2 * p + 0];
    const float oy = ao[2 * p + 1] + boff[h * (NPP * 2) + 2 * p + 1];
    // loc = rp + off/W; grid x = loc*W - 0.5 = rp*W + off - 0.5  (W = H = 128)
    sloc[(size_t)idx * 4 + p] = make_float2(rx * (float)HWW + ox - 0.5f,
                                            ry * (float)HWW + oy - 0.5f);
    awout[(size_t)idx * 4 + p] = lg[p] * inv;
  }
}

// ---------------- kernel 5: bilinear sampling, 32 lanes per (q,h), 4 channels/lane ----------------
__global__ void __launch_bounds__(256)
sample_kernel(const _Float16* __restrict__ value, const float2* __restrict__ sloc,
              const float* __restrict__ aw, _Float16* __restrict__ samp) {
  const int tid  = blockIdx.x * 256 + threadIdx.x;
  const int lane = tid & 31;
  const int idx  = tid >> 5;                       // (b*NQ+q)*NH + h
  if (idx >= ROWS * NHH) return;
  const int h  = idx % NHH;
  const int bq = idx / NHH;
  const int b  = bq / NQ;
  const int d0 = lane * 4;
  const _Float16* vb = value + ((size_t)b * NQ) * CC + h * DD + d0;
  float acc0 = 0.f, acc1 = 0.f, acc2 = 0.f, acc3 = 0.f;
#pragma unroll
  for (int p = 0; p < 4; ++p) {
    const float2 s = sloc[(size_t)idx * 4 + p];
    const float  w = aw[(size_t)idx * 4 + p];
    const float x0f = floorf(s.x), y0f = floorf(s.y);
    const int   x0  = (int)x0f,    y0  = (int)y0f;
    const float wx1 = s.x - x0f, wx0 = 1.f - wx1;
    const float wy1 = s.y - y0f, wy0 = 1.f - wy1;
    const int   xs[4] = { x0, x0 + 1, x0,     x0 + 1 };
    const int   ys[4] = { y0, y0,     y0 + 1, y0 + 1 };
    const float cw[4] = { wx0 * wy0, wx1 * wy0, wx0 * wy1, wx1 * wy1 };
#pragma unroll
    for (int cidx = 0; cidx < 4; ++cidx) {
      const int xi = xs[cidx], yi = ys[cidx];
      if (xi >= 0 && xi < HWW && yi >= 0 && yi < HWW) {
        const v4h v = *(const v4h*)(vb + (size_t)(yi * HWW + xi) * CC);
        const float ww = cw[cidx] * w;
        acc0 += ww * (float)v[0]; acc1 += ww * (float)v[1];
        acc2 += ww * (float)v[2]; acc3 += ww * (float)v[3];
      }
    }
  }
  v4h o; o[0] = (_Float16)acc0; o[1] = (_Float16)acc1;
         o[2] = (_Float16)acc2; o[3] = (_Float16)acc3;
  *(v4h*)(samp + (size_t)bq * CC + h * DD + d0) = o;
}

// ---------------- kernel 6: WMMA GEMM + fused residual epilogue (f32 out) ----------------
__global__ void __launch_bounds__(256)
gemm_final_kernel(const _Float16* __restrict__ A, const _Float16* __restrict__ Bt,
                  const float* __restrict__ bias, const float* __restrict__ query,
                  const float* __restrict__ gamma, float* __restrict__ out,
                  int K, int N) {
  __shared__ _Float16 lbs[2][BN * LDSS];
  const int lane  = threadIdx.x & 31;
  const int wave  = threadIdx.x >> 5;
  const int wm    = wave >> 1, wn = wave & 1;
  const int mrow  = blockIdx.x * BM + wm * 32;
  const int ncol0 = blockIdx.y * BN;
  const int nloc  = wn * 64;

  v8f acc[2][4] = {};
  async_tile_copy(Bt, K, ncol0, 0, lbs[0]);
  const int KT = K / BK;
  for (int t = 0; t < KT; ++t) {
    wait_async();
    __syncthreads();
    if (t + 1 < KT) async_tile_copy(Bt, K, ncol0, (t + 1) * BK, lbs[(t + 1) & 1]);
    const _Float16* lb = lbs[t & 1];
    const int k0 = t * BK;
    v16h a0 = load_a_frag(A, K, mrow,      k0, lane);
    v16h a1 = load_a_frag(A, K, mrow + 16, k0, lane);
#pragma unroll
    for (int j = 0; j < 4; ++j) {
      v16h b = lds_b_frag(lb, nloc + j * 16, lane);
      acc[0][j] = WMMA_F16(a0, b, acc[0][j]);
      acc[1][j] = WMMA_F16(a1, b, acc[1][j]);
    }
  }
  const int c = lane & 15;
#pragma unroll
  for (int i = 0; i < 2; ++i) {
    const int rbase = mrow + i * 16 + ((lane >> 4) << 3);
#pragma unroll
    for (int j = 0; j < 4; ++j) {
      const int col = ncol0 + nloc + j * 16 + c;
      const float bs = bias[col];
      const float gm = gamma[col];
#pragma unroll
      for (int r = 0; r < 8; ++r) {
        const size_t ofs = (size_t)(rbase + r) * N + col;
        out[ofs] = query[ofs] + gm * (acc[i][j][r] + bs);
      }
    }
  }
}

// ---------------- host-side launcher ----------------
extern "C" void kernel_launch(void* const* d_in, const int* in_sizes, int n_in,
                              void* d_out, int out_size, void* d_ws, size_t ws_size,
                              hipStream_t stream) {
  const float* query  = (const float*)d_in[0];
  const float* refpts = (const float*)d_in[1];
  const float* feat   = (const float*)d_in[2];
  // d_in[3] spatial_shapes = [[128,128]], d_in[4] level_start_index = [0] (hardcoded)
  const float* qn_g   = (const float*)d_in[5];
  const float* qn_b   = (const float*)d_in[6];
  const float* fn_g   = (const float*)d_in[7];
  const float* fn_b   = (const float*)d_in[8];
  const float* gamma  = (const float*)d_in[9];
  const float* W_off  = (const float*)d_in[10];
  const float* b_off  = (const float*)d_in[11];
  const float* W_attn = (const float*)d_in[12];
  const float* b_attn = (const float*)d_in[13];
  const float* W_val  = (const float*)d_in[14];
  const float* b_val  = (const float*)d_in[15];
  const float* W_out  = (const float*)d_in[16];
  const float* b_out  = (const float*)d_in[17];
  float* out = (float*)d_out;

  // workspace carving (256B aligned)
  char* ws = (char*)d_ws;
  size_t off = 0;
  auto carve = [&](size_t bytes) -> char* {
    char* p = ws + off;
    off = (off + bytes + 255) & ~(size_t)255;
    return p;
  };
  const size_t actBytes = (size_t)ROWS * CC * sizeof(_Float16);   // 50.3 MB
  _Float16* q16   = (_Float16*)carve(actBytes);    // reused as samp buffer later
  _Float16* f16   = (_Float16*)carve(actBytes);
  _Float16* val16 = (_Float16*)carve(actBytes);
  _Float16* Wv_t  = (_Float16*)carve((size_t)CC * CC * sizeof(_Float16));
  _Float16* Wo_t  = (_Float16*)carve((size_t)CC * CC * sizeof(_Float16));
  float2*   sloc  = (float2*)carve((size_t)ROWS * NHH * NPP * sizeof(float2));
  float*    awbuf = (float*)carve((size_t)ROWS * NHH * NPP * sizeof(float));
  _Float16* samp16 = q16;   // q16 dead after offset_attn_kernel

  // 1) weights -> f16, N-major
  {
    int n = CC * CC;
    wt_convert_kernel<<<(n + 255) / 256, 256, 0, stream>>>(W_val, Wv_t, CC, CC);
    wt_convert_kernel<<<(n + 255) / 256, 256, 0, stream>>>(W_out, Wo_t, CC, CC);
  }
  // 2) layernorms
  layernorm_kernel<<<ROWS, 256, 0, stream>>>(query, qn_g, qn_b, q16);
  layernorm_kernel<<<ROWS, 256, 0, stream>>>(feat,  fn_g, fn_b, f16);
  // 3) value = f_ln @ W_val + b_val   (WMMA, async-LDS staged B)
  {
    dim3 grid(ROWS / BM, CC / BN);
    gemm_bias_f16_kernel<<<grid, 256, 0, stream>>>(f16, Wv_t, b_val, val16, CC, CC);
  }
  // 4) offsets / attention weights / sample locations
  {
    int n = ROWS * NHH;
    offset_attn_kernel<<<(n + 255) / 256, 256, 0, stream>>>(
        q16, refpts, W_off, b_off, W_attn, b_attn, sloc, awbuf);
  }
  // 5) bilinear sampling -> samp (aliases q16)
  {
    int n = ROWS * NHH * 32;
    sample_kernel<<<(n + 255) / 256, 256, 0, stream>>>(val16, sloc, awbuf, samp16);
  }
  // 6) out = query + gamma * (samp @ W_out + b_out)   (WMMA + fused epilogue)
  {
    dim3 grid(ROWS / BM, CC / BN);
    gemm_final_kernel<<<grid, 256, 0, stream>>>(samp16, Wo_t, b_out, query, gamma,
                                                out, CC, CC);
  }
  (void)in_sizes; (void)n_in; (void)out_size; (void)ws_size;
}
